// MAEEGClassification_21088289423845
// MI455X (gfx1250) — compile-verified, matrix operand order checked by reading, not run on hardware
//
#include <hip/hip_runtime.h>
#include <math.h>

// ---------------------------------------------------------------------------
// MI455X (gfx1250, wave32) implementation.
// bf16 WMMA (v_wmma_f32_16x16x32_bf16) for all conv / transformer GEMMs,
// f32 accumulate, f32 residual stream + norms/softmax.
// Each wave register-blocks a 32x32 output tile (4 WMMA accumulators), so
// every loaded A/B fragment feeds two WMMAs (16 FLOP/byte from cache).
// ---------------------------------------------------------------------------

typedef unsigned short u16;
typedef __attribute__((ext_vector_type(16))) __bf16 v16bf;
typedef __attribute__((ext_vector_type(8)))  float  v8f;
typedef __attribute__((ext_vector_type(4)))  unsigned int u32x4;

union Frag { v16bf v; u32x4 q[2]; };

static __device__ __forceinline__ u16 f2b(float f) {
    unsigned u = __float_as_uint(f);
    u += 0x7FFFu + ((u >> 16) & 1u);            // round-to-nearest-even
    return (u16)(u >> 16);
}
static __device__ __forceinline__ float b2f(u16 h) {
    return __uint_as_float(((unsigned)h) << 16);
}
static __device__ __forceinline__ float gelu_exact(float x) {
    return 0.5f * x * (1.f + erff(x * 0.70710678118654752f));
}

// Load a 16x32 bf16 WMMA operand fragment. `p` points at (row, k0 + half*8).
// v0-3 <- 8 bf16 at that offset, v4-7 <- 8 bf16 at +16 elements (ISA 7.12.2).
static __device__ __forceinline__ void load_frag(Frag& f, const u16* p) {
    const u32x4* q = reinterpret_cast<const u32x4*>(p);
    f.q[0] = q[0];
    f.q[1] = q[2];
}

#define WMMA_BF16(A_, B_, C_) \
    __builtin_amdgcn_wmma_f32_16x16x32_bf16(false, (A_).v, false, (B_).v, \
                                            (short)0, (C_), false, false)

// ---------------------------------------------------------------------------
// Generic GEMM:  D(MxN) = A(MxK) * B^T   with A:[M][K] bf16, B:[N][K] bf16.
// grid = (M/128, N/64), block = 256 (8 waves, 4(M) x 2(N), 32x32 per wave).
// Epilogue modes:
//   0: outF[m*ldc+n] = v (+bias[n])
//   1: outBf[m*ldc+n] = bf16(v)
//   2: outBf[m*ldc+n] = bf16(relu(v + bias[n]))
//   3: q/k layout   : outBf[((b*8+h)*512+s)*32+d],  m=(b,s), n=(h,d)
//   4: v^T layout   : outBf[((b*8+h)*32+d)*512+s]
// ---------------------------------------------------------------------------
__global__ __launch_bounds__(256)
void gemm_bf16(const u16* __restrict__ A, const u16* __restrict__ Bm,
               const float* __restrict__ bias,
               float* __restrict__ outF, u16* __restrict__ outBf,
               int K, int ldc, int mode)
{
    const int lane = threadIdx.x & 31;
    const int w    = threadIdx.x >> 5;
    const int m0   = blockIdx.x * 128 + (w & 3) * 32;
    const int n0   = blockIdx.y * 64  + (w >> 2) * 32;
    const int half = lane >> 4;
    const int lr   = lane & 15;

    const u16* pa0 = A  + (size_t)(m0 + lr) * K + half * 8;
    const u16* pa1 = pa0 + (size_t)16 * K;
    const u16* pb0 = Bm + (size_t)(n0 + lr) * K + half * 8;
    const u16* pb1 = pb0 + (size_t)16 * K;

    v8f acc[2][2] = {};
    for (int k = 0; k < K; k += 32) {
        Frag fa0, fa1, fb0, fb1;
        load_frag(fa0, pa0 + k);
        load_frag(fa1, pa1 + k);
        load_frag(fb0, pb0 + k);
        load_frag(fb1, pb1 + k);
        acc[0][0] = WMMA_BF16(fa0, fb0, acc[0][0]);
        acc[0][1] = WMMA_BF16(fa0, fb1, acc[0][1]);
        acc[1][0] = WMMA_BF16(fa1, fb0, acc[1][0]);
        acc[1][1] = WMMA_BF16(fa1, fb1, acc[1][1]);
    }

#pragma unroll
    for (int i = 0; i < 2; ++i) {
#pragma unroll
        for (int j = 0; j < 2; ++j) {
            const int n = n0 + 16 * j + lr;
            const float bv = bias ? bias[n] : 0.f;
#pragma unroll
            for (int r = 0; r < 8; ++r) {
                const int m = m0 + 16 * i + r + half * 8;
                float v = acc[i][j][r] + bv;
                if (mode == 0) {
                    outF[(size_t)m * ldc + n] = v;
                } else if (mode == 1) {
                    outBf[(size_t)m * ldc + n] = f2b(v);
                } else if (mode == 2) {
                    outBf[(size_t)m * ldc + n] = f2b(v > 0.f ? v : 0.f);
                } else if (mode == 3) {
                    const int b = m >> 9, s = m & 511;
                    outBf[((((size_t)b * 8 + (n >> 5)) * 512 + s) << 5) + (n & 31)] = f2b(v);
                } else { // 4
                    const int b = m >> 9, s = m & 511;
                    outBf[(((size_t)b * 8 + (n >> 5)) * 32 + (n & 31)) * 512 + s] = f2b(v);
                }
            }
        }
    }
}

// ---------------------------------------------------------------------------
// Attention scores for one batch, all heads:  sc[h] = Q_h * K_h^T  (K=32).
// q/k: [8][512][32] bf16 (one batch slice), sc: [8][512][512] f32.
// grid = (512/128, 512/64, 8)
// ---------------------------------------------------------------------------
__global__ __launch_bounds__(256)
void attn_scores(const u16* __restrict__ qb, const u16* __restrict__ kb,
                 float* __restrict__ scb)
{
    const int lane = threadIdx.x & 31;
    const int w    = threadIdx.x >> 5;
    const int hh   = blockIdx.z;
    const int m0   = blockIdx.x * 128 + (w & 3) * 32;
    const int n0   = blockIdx.y * 64  + (w >> 2) * 32;
    const int half = lane >> 4;
    const int lr   = lane & 15;

    const u16* q = qb + (size_t)hh * 512 * 32;
    const u16* k = kb + (size_t)hh * 512 * 32;

    Frag fa0, fa1, fb0, fb1;
    load_frag(fa0, q + (size_t)(m0 + lr) * 32 + half * 8);
    load_frag(fa1, q + (size_t)(m0 + 16 + lr) * 32 + half * 8);
    load_frag(fb0, k + (size_t)(n0 + lr) * 32 + half * 8);
    load_frag(fb1, k + (size_t)(n0 + 16 + lr) * 32 + half * 8);

    v8f acc[2][2] = {};
    acc[0][0] = WMMA_BF16(fa0, fb0, acc[0][0]);
    acc[0][1] = WMMA_BF16(fa0, fb1, acc[0][1]);
    acc[1][0] = WMMA_BF16(fa1, fb0, acc[1][0]);
    acc[1][1] = WMMA_BF16(fa1, fb1, acc[1][1]);

    float* o = scb + (size_t)hh * 512 * 512;
#pragma unroll
    for (int i = 0; i < 2; ++i)
#pragma unroll
        for (int j = 0; j < 2; ++j) {
            const int n = n0 + 16 * j + lr;
#pragma unroll
            for (int r = 0; r < 8; ++r) {
                const int m = m0 + 16 * i + r + half * 8;
                o[(size_t)m * 512 + n] = acc[i][j][r];
            }
        }
}

// ---------------------------------------------------------------------------
// Attention * V for one batch, all heads:  o[s, h*32+d] = P_h[s,:] . V_h[:,d].
// Each 32-wide N wave tile lies inside one head (n0 is 32-aligned).
// pr: [8][512][512] bf16, vT: [8][32][512] bf16 (one batch), o: [512][256] bf16
// grid = (512/128, 256/64)
// ---------------------------------------------------------------------------
__global__ __launch_bounds__(256)
void attn_av(const u16* __restrict__ prb, const u16* __restrict__ vTb,
             u16* __restrict__ ob)
{
    const int lane = threadIdx.x & 31;
    const int w    = threadIdx.x >> 5;
    const int m0   = blockIdx.x * 128 + (w & 3) * 32;
    const int n0g  = blockIdx.y * 64  + (w >> 2) * 32;  // global col, 32-aligned
    const int hh   = n0g >> 5;
    const int half = lane >> 4;
    const int lr   = lane & 15;

    const u16* A  = prb + (size_t)hh * 512 * 512;   // [512][512]
    const u16* Bv = vTb + (size_t)hh * 32 * 512;    // [32][512], rows = d

    const u16* pa0 = A + (size_t)(m0 + lr) * 512 + half * 8;
    const u16* pa1 = pa0 + (size_t)16 * 512;
    const u16* pb0 = Bv + (size_t)lr * 512 + half * 8;
    const u16* pb1 = pb0 + (size_t)16 * 512;

    v8f acc[2][2] = {};
    for (int k = 0; k < 512; k += 32) {
        Frag fa0, fa1, fb0, fb1;
        load_frag(fa0, pa0 + k);
        load_frag(fa1, pa1 + k);
        load_frag(fb0, pb0 + k);
        load_frag(fb1, pb1 + k);
        acc[0][0] = WMMA_BF16(fa0, fb0, acc[0][0]);
        acc[0][1] = WMMA_BF16(fa0, fb1, acc[0][1]);
        acc[1][0] = WMMA_BF16(fa1, fb0, acc[1][0]);
        acc[1][1] = WMMA_BF16(fa1, fb1, acc[1][1]);
    }

#pragma unroll
    for (int i = 0; i < 2; ++i)
#pragma unroll
        for (int j = 0; j < 2; ++j) {
            const int n = n0g + 16 * j + lr;
#pragma unroll
            for (int r = 0; r < 8; ++r) {
                const int m = m0 + 16 * i + r + half * 8;
                ob[(size_t)m * 256 + n] = f2b(acc[i][j][r]);
            }
        }
}

// ---------------------------------------------------------------------------
// Implicit-GEMM conv1d (stride 2, pad 7, 15 taps), E=256 out channels.
// Xp  : [B][Tpad][Cin] bf16, data rows at t+8 (8 zero pad rows each side)
// Wp  : [15][256][Cin]  bf16
// out : either bf16 [B][Tout+16][256] (row t+8)  or f32 [B][Tout][256]
// grid = (E/128, Tout/64, B), 32x32 wave tiles.
// ---------------------------------------------------------------------------
__global__ __launch_bounds__(256)
void conv_gemm(const u16* __restrict__ Xp, const u16* __restrict__ Wp,
               const float* __restrict__ bias,
               u16* __restrict__ outBf, float* __restrict__ outF,
               int Cin, int Tpad, int Tout)
{
    const int lane = threadIdx.x & 31;
    const int w    = threadIdx.x >> 5;
    const int m0   = blockIdx.x * 128 + (w & 3) * 32;   // E
    const int n0   = blockIdx.y * 64  + (w >> 2) * 32;  // T
    const int b    = blockIdx.z;
    const int half = lane >> 4;
    const int lr   = lane & 15;

    v8f acc[2][2] = {};
    for (int k = 0; k < 15; ++k) {
        const u16* pa0 = Wp + ((size_t)(k * 256 + m0 + lr)) * Cin + half * 8;
        const u16* pa1 = pa0 + (size_t)16 * Cin;
        const int row0 = 2 * (n0 + lr) + k + 1;        // 2t+k-7 shifted by pad 8
        const u16* pb0 = Xp + ((size_t)b * Tpad + row0) * Cin + half * 8;
        const u16* pb1 = pb0 + (size_t)32 * Cin;       // t+16 -> row += 32
        for (int c = 0; c < Cin; c += 32) {
            Frag fa0, fa1, fb0, fb1;
            load_frag(fa0, pa0 + c);
            load_frag(fa1, pa1 + c);
            load_frag(fb0, pb0 + c);
            load_frag(fb1, pb1 + c);
            acc[0][0] = WMMA_BF16(fa0, fb0, acc[0][0]);
            acc[0][1] = WMMA_BF16(fa0, fb1, acc[0][1]);
            acc[1][0] = WMMA_BF16(fa1, fb0, acc[1][0]);
            acc[1][1] = WMMA_BF16(fa1, fb1, acc[1][1]);
        }
    }

#pragma unroll
    for (int i = 0; i < 2; ++i)
#pragma unroll
        for (int j = 0; j < 2; ++j) {
            const int t = n0 + 16 * j + lr;
#pragma unroll
            for (int r = 0; r < 8; ++r) {
                const int e = m0 + 16 * i + r + half * 8;
                const float v = acc[i][j][r] + bias[e];
                if (outBf) outBf[((size_t)b * (Tout + 16) + t + 8) * 256 + e] = f2b(v);
                else       outF [((size_t)b * Tout + t) * 256 + e] = v;
            }
        }
}

// ---------------------------------------------------------------------------
// GroupNorm (2 channels per group) + exact GELU, in place.
// ---------------------------------------------------------------------------
__global__ __launch_bounds__(256)
void gn_gelu_bf(u16* __restrict__ buf, const float* __restrict__ g,
                const float* __restrict__ bb, int Tout, int Tpad)
{
    __shared__ float s1[256], s2[256];
    const int b   = blockIdx.x >> 7;
    const int grp = blockIdx.x & 127;
    const int c0  = grp * 2;
    const size_t base = ((size_t)b * Tpad + 8) * 256;

    float sum = 0.f, sq = 0.f;
    for (int t = threadIdx.x; t < Tout; t += 256) {
        const float v0 = b2f(buf[base + (size_t)t * 256 + c0]);
        const float v1 = b2f(buf[base + (size_t)t * 256 + c0 + 1]);
        sum += v0 + v1; sq += v0 * v0 + v1 * v1;
    }
    s1[threadIdx.x] = sum; s2[threadIdx.x] = sq; __syncthreads();
    for (int o = 128; o > 0; o >>= 1) {
        if ((int)threadIdx.x < o) { s1[threadIdx.x] += s1[threadIdx.x + o];
                                    s2[threadIdx.x] += s2[threadIdx.x + o]; }
        __syncthreads();
    }
    const float inv_n = 1.f / (2.f * (float)Tout);
    const float mu   = s1[0] * inv_n;
    const float var  = s2[0] * inv_n - mu * mu;
    const float rstd = rsqrtf(var + 1e-5f);
    const float g0 = g[c0], g1 = g[c0 + 1], b0 = bb[c0], b1 = bb[c0 + 1];
    for (int t = threadIdx.x; t < Tout; t += 256) {
        const size_t i0 = base + (size_t)t * 256 + c0;
        const float v0 = (b2f(buf[i0])     - mu) * rstd * g0 + b0;
        const float v1 = (b2f(buf[i0 + 1]) - mu) * rstd * g1 + b1;
        buf[i0]     = f2b(gelu_exact(v0));
        buf[i0 + 1] = f2b(gelu_exact(v1));
    }
}

__global__ __launch_bounds__(256)
void gn_gelu_f32(float* __restrict__ buf, const float* __restrict__ g,
                 const float* __restrict__ bb)
{
    __shared__ float s1[256], s2[256];
    const int b   = blockIdx.x >> 7;
    const int grp = blockIdx.x & 127;
    const int c0  = grp * 2;
    const size_t base = (size_t)b * 512 * 256;

    float sum = 0.f, sq = 0.f;
    for (int t = threadIdx.x; t < 512; t += 256) {
        const float v0 = buf[base + (size_t)t * 256 + c0];
        const float v1 = buf[base + (size_t)t * 256 + c0 + 1];
        sum += v0 + v1; sq += v0 * v0 + v1 * v1;
    }
    s1[threadIdx.x] = sum; s2[threadIdx.x] = sq; __syncthreads();
    for (int o = 128; o > 0; o >>= 1) {
        if ((int)threadIdx.x < o) { s1[threadIdx.x] += s1[threadIdx.x + o];
                                    s2[threadIdx.x] += s2[threadIdx.x + o]; }
        __syncthreads();
    }
    const float inv_n = 1.f / 1024.f;
    const float mu   = s1[0] * inv_n;
    const float var  = s2[0] * inv_n - mu * mu;
    const float rstd = rsqrtf(var + 1e-5f);
    const float g0 = g[c0], g1 = g[c0 + 1], b0 = bb[c0], b1 = bb[c0 + 1];
    for (int t = threadIdx.x; t < 512; t += 256) {
        const size_t i0 = base + (size_t)t * 256 + c0;
        buf[i0]     = gelu_exact((buf[i0]     - mu) * rstd * g0 + b0);
        buf[i0 + 1] = gelu_exact((buf[i0 + 1] - mu) * rstd * g1 + b1);
    }
}

// LayerNorm(a + h) -> h (f32) and h_bf (bf16). grid=16384 rows, block=256.
__global__ __launch_bounds__(256)
void ln_add(const float* __restrict__ a, float* __restrict__ h,
            u16* __restrict__ hbf, const float* __restrict__ g,
            const float* __restrict__ bb)
{
    __shared__ float s1[256], s2[256];
    const size_t row = blockIdx.x;
    const int e = threadIdx.x;
    const float v = a[row * 256 + e] + h[row * 256 + e];
    s1[e] = v; s2[e] = v * v; __syncthreads();
    for (int o = 128; o > 0; o >>= 1) {
        if (e < o) { s1[e] += s1[e + o]; s2[e] += s2[e + o]; }
        __syncthreads();
    }
    const float mu   = s1[0] * (1.f / 256.f);
    const float var  = s2[0] * (1.f / 256.f) - mu * mu;
    const float rstd = rsqrtf(var + 1e-5f);
    const float y = (v - mu) * rstd * g[e] + bb[e];
    h[row * 256 + e]   = y;
    hbf[row * 256 + e] = f2b(y);
}

// h += pos_encoding(s, e);  also refresh bf16 copy. grid=16384, block=256.
__global__ __launch_bounds__(256)
void pos_add(float* __restrict__ h, u16* __restrict__ hbf)
{
    const size_t row = blockIdx.x;
    const int e = threadIdx.x;
    const int s = (int)(row & 511);
    // 10000^(e/256) -> exp2(e * log2(10000)/256)
    const float ang = (float)s * exp2f(-(float)e * (13.287712379549449f / 256.f));
    const float p = ((e & 1) == 0) ? sinf(ang) : cosf(ang);
    const float y = h[row * 256 + e] + p;
    h[row * 256 + e]   = y;
    hbf[row * 256 + e] = f2b(y);
}

// Row softmax over 512 scores (one batch, 8 heads) -> bf16 probs.
__global__ __launch_bounds__(256)
void softmax_bf(const float* __restrict__ sc, u16* __restrict__ pr)
{
    __shared__ float red[256];
    const size_t row = blockIdx.x;        // 8*512 rows
    const float* x = sc + row * 512;
    const int t = threadIdx.x;
    const float scale = 0.17677669529663687f;  // 1/sqrt(32)
    const float a0 = x[t] * scale, a1 = x[t + 256] * scale;
    red[t] = fmaxf(a0, a1); __syncthreads();
    for (int o = 128; o > 0; o >>= 1) {
        if (t < o) red[t] = fmaxf(red[t], red[t + o]);
        __syncthreads();
    }
    const float mx = red[0]; __syncthreads();
    const float e0 = expf(a0 - mx), e1 = expf(a1 - mx);
    red[t] = e0 + e1; __syncthreads();
    for (int o = 128; o > 0; o >>= 1) {
        if (t < o) red[t] += red[t + o];
        __syncthreads();
    }
    const float inv = 1.f / red[0];
    pr[row * 512 + t]       = f2b(e0 * inv);
    pr[row * 512 + t + 256] = f2b(e1 * inv);
}

// Classifier stage 1: hid[b][n] = relu(dot(h_flat[b], Wc1[n]) + bc1[n]).
__global__ __launch_bounds__(256)
void cls1(const float* __restrict__ h, const float* __restrict__ Wc1,
          const float* __restrict__ bc1, float* __restrict__ hid)
{
    __shared__ float red[256];
    const int b = blockIdx.x >> 8;
    const int n = blockIdx.x & 255;
    const float* hr = h   + (size_t)b * 131072;
    const float* wr = Wc1 + (size_t)n * 131072;
    float s = 0.f;
    for (int i = threadIdx.x; i < 131072; i += 256) s += hr[i] * wr[i];
    red[threadIdx.x] = s; __syncthreads();
    for (int o = 128; o > 0; o >>= 1) {
        if ((int)threadIdx.x < o) red[threadIdx.x] += red[threadIdx.x + o];
        __syncthreads();
    }
    if (threadIdx.x == 0) {
        const float v = red[0] + bc1[n];
        hid[b * 256 + n] = v > 0.f ? v : 0.f;
    }
}

// Classifier stage 2: out[b] = sigmoid(dot(hid[b], Wc2) + bc2).
__global__ __launch_bounds__(256)
void cls2(const float* __restrict__ hid, const float* __restrict__ Wc2,
          const float* __restrict__ bc2, float* __restrict__ out)
{
    __shared__ float red[256];
    const int b = blockIdx.x;
    red[threadIdx.x] = hid[b * 256 + threadIdx.x] * Wc2[threadIdx.x];
    __syncthreads();
    for (int o = 128; o > 0; o >>= 1) {
        if ((int)threadIdx.x < o) red[threadIdx.x] += red[threadIdx.x + o];
        __syncthreads();
    }
    if (threadIdx.x == 0)
        out[b] = 1.f / (1.f + expf(-(red[0] + bc2[0])));
}

// ------------------------- packing / conversion ----------------------------
__global__ __launch_bounds__(256)
void cvt_bf16(const float* __restrict__ src, u16* __restrict__ dst, int n)
{
    const int i = blockIdx.x * 256 + threadIdx.x;
    if (i < n) dst[i] = f2b(src[i]);
}

// x[32][64][4096] f32 -> xT[32][4112][64] bf16 (row t+8)
__global__ __launch_bounds__(256)
void pack_x(const float* __restrict__ x, u16* __restrict__ xp)
{
    const int i = blockIdx.x * 256 + threadIdx.x;   // 2^23 elements
    const int t  = i & 4095;
    const int ci = (i >> 12) & 63;
    const int b  = i >> 18;
    xp[((size_t)b * 4112 + t + 8) * 64 + ci] = f2b(x[i]);
}

// conv0_w[256][64][15] -> [15][256][64]
__global__ __launch_bounds__(256)
void pack_w0(const float* __restrict__ w, u16* __restrict__ wp)
{
    const int i = blockIdx.x * 256 + threadIdx.x;   // 245760
    const int k  = i % 15;
    const int ci = (i / 15) % 64;
    const int e  = i / (15 * 64);
    wp[((size_t)k * 256 + e) * 64 + ci] = f2b(w[i]);
}

// conv12_w[2][256][256][15] -> [2][15][256][256]
__global__ __launch_bounds__(256)
void pack_w12(const float* __restrict__ w, u16* __restrict__ wp)
{
    const int i = blockIdx.x * 256 + threadIdx.x;   // 1966080
    const int k  = i % 15;
    const int ci = (i / 15) % 256;
    const int e  = (i / (15 * 256)) % 256;
    const int j  = i / (15 * 256 * 256);
    wp[(((size_t)j * 15 + k) * 256 + e) * 256 + ci] = f2b(w[i]);
}

// ---------------------------------------------------------------------------
// Workspace layout (bytes)
// ---------------------------------------------------------------------------
static constexpr size_t SZ_XT  = 32ull * 4112 * 64 * 2;     // x transposed+pad
static constexpr size_t SZ_W0  = 15ull * 256 * 64 * 2;      // conv0 weights
static constexpr size_t SZ_W12 = 2ull * 15 * 256 * 256 * 2; // conv1/2 weights
static constexpr size_t SZ_H0  = 32ull * 2064 * 256 * 2;    // conv0 out (pad)
static constexpr size_t SZ_H1  = 32ull * 1040 * 256 * 2;    // conv1 out (pad)
static constexpr size_t SZ_H   = 32ull * 512 * 256 * 4;     // residual stream
static constexpr size_t SZ_HBF = 32ull * 512 * 256 * 2;     // bf16 copy of h
static constexpr size_t SZ_WL  = 4ull * 256 * 256 * 2;      // per Wq/Wk/Wv/Wo
static constexpr size_t SZ_WF  = 4ull * 1024 * 256 * 2;     // per Wf1/Wf2
static constexpr size_t SZ_QKV = 32ull * 8 * 512 * 32 * 2;  // per q/k/vT
static constexpr size_t SZ_SC  = 8ull * 512 * 512 * 4;      // scores (1 batch)
static constexpr size_t SZ_PR  = 8ull * 512 * 512 * 2;      // probs  (1 batch)
static constexpr size_t SZ_OBF = 32ull * 512 * 256 * 2;     // attn out
static constexpr size_t SZ_TMP = 32ull * 512 * 256 * 4;     // f32 GEMM out
static constexpr size_t SZ_FBF = 32ull * 512 * 1024 * 2;    // FFN hidden
static constexpr size_t SZ_HID = 32ull * 256 * 4;           // classifier hid

extern "C" void kernel_launch(void* const* d_in, const int* in_sizes, int n_in,
                              void* d_out, int out_size, void* d_ws, size_t ws_size,
                              hipStream_t stream)
{
    (void)in_sizes; (void)n_in; (void)out_size; (void)ws_size;

    const float* x        = (const float*)d_in[0];
    const float* conv0_w  = (const float*)d_in[1];
    const float* conv0_b  = (const float*)d_in[2];
    const float* conv12_w = (const float*)d_in[3];
    const float* conv12_b = (const float*)d_in[4];
    const float* gn_g     = (const float*)d_in[5];
    const float* gn_b     = (const float*)d_in[6];
    const float* Wq       = (const float*)d_in[7];
    const float* Wk       = (const float*)d_in[8];
    const float* Wv       = (const float*)d_in[9];
    const float* Wo       = (const float*)d_in[10];
    const float* ln1_g    = (const float*)d_in[11];
    const float* ln1_b    = (const float*)d_in[12];
    const float* Wf1      = (const float*)d_in[13];
    const float* bf1      = (const float*)d_in[14];
    const float* Wf2      = (const float*)d_in[15];
    const float* bf2      = (const float*)d_in[16];
    const float* ln2_g    = (const float*)d_in[17];
    const float* ln2_b    = (const float*)d_in[18];
    const float* Wc1      = (const float*)d_in[19];
    const float* bc1      = (const float*)d_in[20];
    const float* Wc2      = (const float*)d_in[21];
    const float* bc2      = (const float*)d_in[22];
    float* out = (float*)d_out;

    char* ws = (char*)d_ws;
    size_t off = 0;
    auto take = [&](size_t sz) { char* p = ws + off; off += sz; return p; };

    u16*   xT    = (u16*)  take(SZ_XT);
    u16*   w0pk  = (u16*)  take(SZ_W0);
    u16*   w12pk = (u16*)  take(SZ_W12);
    u16*   h0pad = (u16*)  take(SZ_H0);
    u16*   h1pad = (u16*)  take(SZ_H1);
    float* h     = (float*)take(SZ_H);
    u16*   hbf   = (u16*)  take(SZ_HBF);
    u16*   WqBf  = (u16*)  take(SZ_WL);
    u16*   WkBf  = (u16*)  take(SZ_WL);
    u16*   WvBf  = (u16*)  take(SZ_WL);
    u16*   WoBf  = (u16*)  take(SZ_WL);
    u16*   Wf1Bf = (u16*)  take(SZ_WF);
    u16*   Wf2Bf = (u16*)  take(SZ_WF);
    u16*   qbf   = (u16*)  take(SZ_QKV);
    u16*   kbf   = (u16*)  take(SZ_QKV);
    u16*   vT    = (u16*)  take(SZ_QKV);
    float* sc    = (float*)take(SZ_SC);
    u16*   pr    = (u16*)  take(SZ_PR);
    u16*   obf   = (u16*)  take(SZ_OBF);
    float* tmp32 = (float*)take(SZ_TMP);
    u16*   fbf   = (u16*)  take(SZ_FBF);
    float* hid   = (float*)take(SZ_HID);

    // Zero workspace (pad rows of conv buffers must be zero). ~200 MB.
    hipMemsetAsync(d_ws, 0, off, stream);

    // --- pack inputs / weights to bf16 ---
    pack_x  <<<32768, 256, 0, stream>>>(x, xT);
    pack_w0 <<<960,   256, 0, stream>>>(conv0_w, w0pk);
    pack_w12<<<7680,  256, 0, stream>>>(conv12_w, w12pk);
    cvt_bf16<<<1024, 256, 0, stream>>>(Wq,  WqBf,  262144);
    cvt_bf16<<<1024, 256, 0, stream>>>(Wk,  WkBf,  262144);
    cvt_bf16<<<1024, 256, 0, stream>>>(Wv,  WvBf,  262144);
    cvt_bf16<<<1024, 256, 0, stream>>>(Wo,  WoBf,  262144);
    cvt_bf16<<<4096, 256, 0, stream>>>(Wf1, Wf1Bf, 1048576);
    cvt_bf16<<<4096, 256, 0, stream>>>(Wf2, Wf2Bf, 1048576);

    // --- conv encoder ---
    conv_gemm<<<dim3(2, 32, 32), 256, 0, stream>>>(
        xT, w0pk, conv0_b, h0pad, nullptr, 64, 4112, 2048);
    gn_gelu_bf<<<4096, 256, 0, stream>>>(h0pad, gn_g + 0,   gn_b + 0,   2048, 2064);

    conv_gemm<<<dim3(2, 16, 32), 256, 0, stream>>>(
        h0pad, w12pk, conv12_b, h1pad, nullptr, 256, 2064, 1024);
    gn_gelu_bf<<<4096, 256, 0, stream>>>(h1pad, gn_g + 256, gn_b + 256, 1024, 1040);

    conv_gemm<<<dim3(2, 8, 32), 256, 0, stream>>>(
        h1pad, w12pk + 15 * 256 * 256, conv12_b + 256, nullptr, h, 256, 1040, 512);
    gn_gelu_f32<<<4096, 256, 0, stream>>>(h, gn_g + 512, gn_b + 512);

    pos_add<<<16384, 256, 0, stream>>>(h, hbf);

    auto gemm = [&](const u16* A, const u16* B, const float* bias,
                    float* oF, u16* oBf, int M, int N, int K, int ldc, int mode) {
        gemm_bf16<<<dim3(M / 128, N / 64), 256, 0, stream>>>(
            A, B, bias, oF, oBf, K, ldc, mode);
    };

    // --- transformer layers ---
    for (int l = 0; l < 4; ++l) {
        const u16* wq = WqBf  + (size_t)l * 65536;
        const u16* wk = WkBf  + (size_t)l * 65536;
        const u16* wv = WvBf  + (size_t)l * 65536;
        const u16* wo = WoBf  + (size_t)l * 65536;
        const u16* w1 = Wf1Bf + (size_t)l * 262144;
        const u16* w2 = Wf2Bf + (size_t)l * 262144;

        gemm(hbf, wq, nullptr, nullptr, qbf, 16384, 256, 256, 0, 3);
        gemm(hbf, wk, nullptr, nullptr, kbf, 16384, 256, 256, 0, 3);
        gemm(hbf, wv, nullptr, nullptr, vT,  16384, 256, 256, 0, 4);

        for (int b = 0; b < 32; ++b) {
            attn_scores<<<dim3(4, 8, 8), 256, 0, stream>>>(
                qbf + (size_t)b * 8 * 512 * 32,
                kbf + (size_t)b * 8 * 512 * 32, sc);
            softmax_bf<<<4096, 256, 0, stream>>>(sc, pr);
            attn_av<<<dim3(4, 4), 256, 0, stream>>>(
                pr, vT + (size_t)b * 8 * 32 * 512,
                obf + (size_t)b * 512 * 256);
        }

        gemm(obf, wo, nullptr, tmp32, nullptr, 16384, 256, 256, 256, 0);
        ln_add<<<16384, 256, 0, stream>>>(tmp32, h, hbf, ln1_g + l * 256, ln1_b + l * 256);

        gemm(hbf, w1, bf1 + l * 1024, nullptr, fbf, 16384, 1024, 256, 1024, 2);
        gemm(fbf, w2, bf2 + l * 256,  tmp32, nullptr, 16384, 256, 1024, 256, 0);
        ln_add<<<16384, 256, 0, stream>>>(tmp32, h, hbf, ln2_g + l * 256, ln2_b + l * 256);
    }

    // --- classifier ---
    cls1<<<32 * 256, 256, 0, stream>>>(h, Wc1, bc1, hid);
    cls2<<<32, 256, 0, stream>>>(hid, Wc2, bc2, out);
}
